// Attention_73203422593796
// MI455X (gfx1250) — matrix operand order repaired and checked by reading, not compiled, not run on hardware
//
#include <hip/hip_runtime.h>
#include <hip/hip_bf16.h>

#define B_   2
#define S_   2048
#define HID_ 2048
#define H_   32
#define HKV_ 8
#define D_   128

typedef __attribute__((ext_vector_type(16))) __bf16 v16bf;
typedef __attribute__((ext_vector_type(8)))  float  v8f;
typedef __attribute__((ext_vector_type(4)))  unsigned int u32x4;
typedef __attribute__((ext_vector_type(8)))  unsigned int u32x8;

union FragB { v16bf v; uint4 q[2]; };
union FragF { v8f  v; float f[8]; };

__device__ __forceinline__ v8f zero_v8f() {
  v8f z;
#pragma unroll
  for (int i = 0; i < 8; ++i) z[i] = 0.0f;
  return z;
}

__device__ __forceinline__ unsigned short f2bf(float f) {
  unsigned int u = __float_as_uint(f);
  u += 0x7FFFu + ((u >> 16) & 1u);   // round-to-nearest-even
  return (unsigned short)(u >> 16);
}
__device__ __forceinline__ float bf2f(unsigned short u) {
  return __uint_as_float(((unsigned int)u) << 16);
}

// A-matrix fragment (16x32 bf16), source row-major with leading dim ld.
// Lanes 0-15: K = 0..7 and 16..23 ; lanes 16-31: K = 8..15 and 24..31.
__device__ __forceinline__ v16bf load_fragA(const unsigned short* __restrict__ base,
                                            int ld, int row, int k0, int lane) {
  const int sub = (lane & 16) ? 8 : 0;
  const unsigned short* p = base + (size_t)row * ld + k0 + sub;
  FragB f;
  f.q[0] = *reinterpret_cast<const uint4*>(p);
  f.q[1] = *reinterpret_cast<const uint4*>(p + 16);
  return f.v;
}

// B-matrix fragment (32x16 bf16) loaded from B^T stored row-major [N x ld].
// Lanes 0-15: K = 0..7 and 16..23 ; lanes 16-31: K = 8..15 and 24..31.
__device__ __forceinline__ v16bf load_fragBt(const unsigned short* __restrict__ bt,
                                             int ld, int n, int k0, int lane) {
  const int sub = (lane & 16) ? 8 : 0;
  const unsigned short* p = bt + (size_t)n * ld + k0 + sub;
  FragB f;
  f.q[0] = *reinterpret_cast<const uint4*>(p);
  f.q[1] = *reinterpret_cast<const uint4*>(p + 16);
  return f.v;
}

// -------------------------------------------------- TDM descriptor helpers
// D# group 0: count=1, lds_addr, global_addr, type=2 ("image")
__device__ __forceinline__ u32x4 tdm_g0(unsigned lds_addr, const void* gptr) {
  const unsigned long long ga = (unsigned long long)gptr;
  u32x4 g;
  g[0] = 1u;                                               // count=1
  g[1] = lds_addr;                                         // bits 63:32
  g[2] = (unsigned)ga;                                     // bits 95:64
  g[3] = ((unsigned)(ga >> 32) & 0x1FFFFFFu) | (2u << 30); // addr[56:32] | type=2
  return g;
}
// D# group 1: 2-D tile, data_size=2B, no multicast/pad/iterate.
__device__ __forceinline__ u32x8 tdm_g1(unsigned tensor_d0, unsigned tensor_d1,
                                        unsigned tile_d0, unsigned tile_d1,
                                        unsigned long long stride0) {
  u32x8 g;
  g[0] = (1u << 16);                                       // data_size = 2 bytes
  g[1] = (tensor_d0 & 0xFFFFu) << 16;                      // tensor_dim0[15:0]
  g[2] = (tensor_d0 >> 16) | ((tensor_d1 & 0xFFFFu) << 16);
  g[3] = (tensor_d1 >> 16) | ((tile_d0 & 0xFFFFu) << 16);
  g[4] = (tile_d1 & 0xFFFFu);                              // tile_dim2 = 0
  g[5] = (unsigned)stride0;                                // tensor_dim0_stride lo
  g[6] = (unsigned)(stride0 >> 32) & 0xFFFFu;              // stride hi | dim1_stride=0
  g[7] = 0u;
  return g;
}

__device__ __forceinline__ void tdm_load(u32x4 g0, u32x8 g1) {
  asm volatile("tensor_load_to_lds %0, %1" :: "s"(g0), "s"(g1) : "memory");
}

// ---------------------------------------------------------------- converts
__global__ void f32_to_bf16_k(const float* __restrict__ in,
                              unsigned short* __restrict__ out, int n) {
  int t = blockIdx.x * blockDim.x + threadIdx.x;
  if (t < n) out[t] = f2bf(in[t]);
}

// wt[n*K + k] = bf16(w[k*N + n])   (store weights transposed, N x K)
__global__ void f32_to_bf16_T(const float* __restrict__ w,
                              unsigned short* __restrict__ wt, int K, int N) {
  int t = blockIdx.x * blockDim.x + threadIdx.x;
  if (t >= K * N) return;
  int n = t % N;
  int k = t / N;
  wt[(size_t)n * K + k] = f2bf(w[t]);
}

// ---------------------------------------------------------------- GEMM (TDM-fed)
// C[M,N] = A[M,K] (bf16 row-major) * B (given as B^T = Bt[N,K] bf16 row-major)
// Workgroup: 8 waves (2x4) -> 64x256 C tile; wave: 32x64 (2x4 WMMA subtiles).
// Per 32-wide K step, wave 0 DMAs the A (64x32) and Bt (256x32) tiles into
// LDS via the Tensor Data Mover, double buffered; all waves consume via LDS.
#define BM 64
#define BN 256
#define BK 32

template <bool OUT_BF16>
__global__ __launch_bounds__(256) void gemm_bf16_tdm(
    const unsigned short* __restrict__ A,
    const unsigned short* __restrict__ Bt,
    void* __restrict__ C, int M, int N, int K) {
  __shared__ __align__(16) unsigned short smem[2][(BM + BN) * BK];  // 2 x 20KB
  const int lane = threadIdx.x & 31;
  const int wid  = threadIdx.x >> 5;          // 0..7
  const int tilesN = N / BN;
  const int blockTm = (blockIdx.x / tilesN) * BM;
  const int blockTn = (blockIdx.x % tilesN) * BN;
  const int tmL = (wid >> 2) * 32;            // 0 or 32
  const int tnL = (wid & 3) * 64;             // 0,64,128,192
  const int col = lane & 15;

  // LDS byte offset of smem (generic LDS pointer: addr[31:0] is the offset).
  const unsigned ldsBase  = (unsigned)(size_t)(const void*)&smem[0][0];
  const unsigned bufBytes = (BM + BN) * BK * 2u;
  const unsigned bOffB    = BM * BK * 2u;     // Bt tile offset inside a buffer

  auto issue_tiles = [&](int kk, int buf) {
    const unsigned lb = ldsBase + (unsigned)buf * bufBytes;
    tdm_load(tdm_g0(lb,         A  + (size_t)blockTm * K + kk),
             tdm_g1(BK, BM, BK, BM, (unsigned long long)K));
    tdm_load(tdm_g0(lb + bOffB, Bt + (size_t)blockTn * K + kk),
             tdm_g1(BK, BN, BK, BN, (unsigned long long)K));
  };

  if (wid == 0) {
    issue_tiles(0, 0);
    __builtin_amdgcn_s_wait_tensorcnt(0);
  }
  __syncthreads();

  v8f acc[2][4];
#pragma unroll
  for (int i = 0; i < 2; ++i)
#pragma unroll
    for (int j = 0; j < 4; ++j) acc[i][j] = zero_v8f();

  const int nk = K / BK;
  int buf = 0;
  for (int i = 0; i < nk; ++i) {
    const bool more = (i + 1) < nk;
    if (wid == 0 && more) issue_tiles((i + 1) * BK, buf ^ 1);

    const unsigned short* sA = &smem[buf][0];
    const unsigned short* sB = &smem[buf][BM * BK];

    v16bf a0 = load_fragA(sA, BK, tmL + col,      0, lane);
    v16bf a1 = load_fragA(sA, BK, tmL + 16 + col, 0, lane);
    v16bf b0 = load_fragBt(sB, BK, tnL + col,      0, lane);
    v16bf b1 = load_fragBt(sB, BK, tnL + 16 + col, 0, lane);
    v16bf b2 = load_fragBt(sB, BK, tnL + 32 + col, 0, lane);
    v16bf b3 = load_fragBt(sB, BK, tnL + 48 + col, 0, lane);

    acc[0][0] = __builtin_amdgcn_wmma_f32_16x16x32_bf16(false, a0, false, b0, (short)0, acc[0][0], false, false);
    acc[0][1] = __builtin_amdgcn_wmma_f32_16x16x32_bf16(false, a0, false, b1, (short)0, acc[0][1], false, false);
    acc[0][2] = __builtin_amdgcn_wmma_f32_16x16x32_bf16(false, a0, false, b2, (short)0, acc[0][2], false, false);
    acc[0][3] = __builtin_amdgcn_wmma_f32_16x16x32_bf16(false, a0, false, b3, (short)0, acc[0][3], false, false);
    acc[1][0] = __builtin_amdgcn_wmma_f32_16x16x32_bf16(false, a1, false, b0, (short)0, acc[1][0], false, false);
    acc[1][1] = __builtin_amdgcn_wmma_f32_16x16x32_bf16(false, a1, false, b1, (short)0, acc[1][1], false, false);
    acc[1][2] = __builtin_amdgcn_wmma_f32_16x16x32_bf16(false, a1, false, b2, (short)0, acc[1][2], false, false);
    acc[1][3] = __builtin_amdgcn_wmma_f32_16x16x32_bf16(false, a1, false, b3, (short)0, acc[1][3], false, false);

    if (wid == 0 && more) __builtin_amdgcn_s_wait_tensorcnt(0);
    __syncthreads();
    buf ^= 1;
  }

  const int hgrp = lane >> 4;
  const int tm = blockTm + tmL;
  const int tn = blockTn + tnL;
#pragma unroll
  for (int i = 0; i < 2; ++i) {
#pragma unroll
    for (int j = 0; j < 4; ++j) {
      FragF cf; cf.v = acc[i][j];
#pragma unroll
      for (int r = 0; r < 8; ++r) {
        const int m = tm + i * 16 + r + hgrp * 8;
        const int n = tn + j * 16 + col;
        if (OUT_BF16)
          ((unsigned short*)C)[(size_t)m * N + n] = f2bf(cf.f[r]);
        else
          ((float*)C)[(size_t)m * N + n] = cf.f[r];
      }
    }
  }
}

// ---------------------------------------------------------------- RoPE (in place, bf16 [B,S,NH,D])
__global__ void rope_inplace(unsigned short* __restrict__ x,
                             const float* __restrict__ cosT,
                             const float* __restrict__ sinT,
                             int NH, int total) {
  int t = blockIdx.x * blockDim.x + threadIdx.x;
  if (t >= total) return;
  const int d  = t & 63;
  const int hh = (t >> 6) % NH;
  const int s  = (t / (64 * NH)) % S_;
  const int b  = t / (64 * NH * S_);
  const size_t base = (((size_t)b * S_ + s) * NH + hh) * D_;
  const float x0 = bf2f(x[base + d]);
  const float x1 = bf2f(x[base + d + 64]);
  const float c  = cosT[(size_t)s * D_ + d];
  const float sn = sinT[(size_t)s * D_ + d];
  x[base + d]      = f2bf(x0 * c - x1 * sn);
  x[base + d + 64] = f2bf(x0 * sn + x1 * c);
}

// V [B,S,HKV,D] -> Vt [B,HKV,D,S]
__global__ void transpose_v(const unsigned short* __restrict__ v,
                            unsigned short* __restrict__ vt, int total) {
  int t = blockIdx.x * blockDim.x + threadIdx.x;
  if (t >= total) return;
  const int s   = t % S_;
  const int d   = (t / S_) % D_;
  const int kvh = (t / (S_ * D_)) % HKV_;
  const int b   = t / (S_ * D_ * HKV_);
  vt[t] = v[(((size_t)b * S_ + s) * HKV_ + kvh) * D_ + d];
}

// ---------------------------------------------------------------- flash attention
// One wave: 16 query rows of one (b,h). Keys processed in 64-wide chunks;
// the causal mask is only applied in the (single) diagonal chunk via the
// MASKED template instantiation, so the steady-state loop carries no
// compare/cndmask work. Softmax runs in the base-2 domain (native v_exp_f32).
template <bool MASKED>
__device__ __forceinline__ void fa_chunk64(
    int kb, int qbase, int lane, const v16bf* qf,
    const unsigned short* __restrict__ Kbase,
    const unsigned short* __restrict__ Vbase,
    unsigned short* lp, FragF* o, float* m_i, float* l_i) {
  const int col  = lane & 15;
  const int hgrp = lane >> 4;
  const float kscale = 0.08838834764831845f * 1.4426950408889634f;  // 1/sqrt(D)*log2(e)

  v8f t0 = zero_v8f(), t1 = zero_v8f(), t2 = zero_v8f(), t3 = zero_v8f();
#pragma unroll
  for (int dd = 0; dd < 4; ++dd) {
    v16bf kf0 = load_fragBt(Kbase, HKV_ * D_, kb + col,      dd * 32, lane);
    v16bf kf1 = load_fragBt(Kbase, HKV_ * D_, kb + 16 + col, dd * 32, lane);
    v16bf kf2 = load_fragBt(Kbase, HKV_ * D_, kb + 32 + col, dd * 32, lane);
    v16bf kf3 = load_fragBt(Kbase, HKV_ * D_, kb + 48 + col, dd * 32, lane);
    t0 = __builtin_amdgcn_wmma_f32_16x16x32_bf16(false, qf[dd], false, kf0, (short)0, t0, false, false);
    t1 = __builtin_amdgcn_wmma_f32_16x16x32_bf16(false, qf[dd], false, kf1, (short)0, t1, false, false);
    t2 = __builtin_amdgcn_wmma_f32_16x16x32_bf16(false, qf[dd], false, kf2, (short)0, t2, false, false);
    t3 = __builtin_amdgcn_wmma_f32_16x16x32_bf16(false, qf[dd], false, kf3, (short)0, t3, false, false);
  }
  FragF s[4];
  s[0].v = t0; s[1].v = t1; s[2].v = t2; s[3].v = t3;

#pragma unroll
  for (int r = 0; r < 8; ++r) {
    float vq[4];
#pragma unroll
    for (int st = 0; st < 4; ++st) vq[st] = s[st].f[r] * kscale;
    if (MASKED) {
      const int qrow = qbase + r + hgrp * 8;
#pragma unroll
      for (int st = 0; st < 4; ++st)
        if (kb + st * 16 + col > qrow) vq[st] = -1e30f;
    }
    float mx = fmaxf(fmaxf(vq[0], vq[1]), fmaxf(vq[2], vq[3]));
    mx = fmaxf(mx, __shfl_xor(mx, 1, 32));
    mx = fmaxf(mx, __shfl_xor(mx, 2, 32));
    mx = fmaxf(mx, __shfl_xor(mx, 4, 32));
    mx = fmaxf(mx, __shfl_xor(mx, 8, 32));
    const float mn   = fmaxf(m_i[r], mx);
    const float corr = exp2f(m_i[r] - mn);
    m_i[r] = mn;
    float p[4], rs = 0.0f;
#pragma unroll
    for (int st = 0; st < 4; ++st) { p[st] = exp2f(vq[st] - mn); rs += p[st]; }
    rs += __shfl_xor(rs, 1, 32);
    rs += __shfl_xor(rs, 2, 32);
    rs += __shfl_xor(rs, 4, 32);
    rs += __shfl_xor(rs, 8, 32);
    l_i[r] = l_i[r] * corr + rs;
#pragma unroll
    for (int dt = 0; dt < 8; ++dt) o[dt].f[r] *= corr;
    const int prow = r + hgrp * 8;
#pragma unroll
    for (int st = 0; st < 4; ++st) lp[prow * 64 + st * 16 + col] = f2bf(p[st]);
  }
  asm volatile("s_wait_dscnt 0" ::: "memory");  // wave-local LDS RAW fence
  const int sub = (lane & 16) ? 8 : 0;
  FragB pf[2];
  pf[0].q[0] = *reinterpret_cast<const uint4*>(&lp[col * 64 + sub]);
  pf[0].q[1] = *reinterpret_cast<const uint4*>(&lp[col * 64 + sub + 16]);
  pf[1].q[0] = *reinterpret_cast<const uint4*>(&lp[col * 64 + 32 + sub]);
  pf[1].q[1] = *reinterpret_cast<const uint4*>(&lp[col * 64 + 32 + sub + 16]);
#pragma unroll
  for (int kh = 0; kh < 2; ++kh) {
#pragma unroll
    for (int dt = 0; dt < 8; ++dt) {
      v16bf vf = load_fragBt(Vbase + kb + kh * 32, S_, dt * 16 + col, 0, lane);
      o[dt].v = __builtin_amdgcn_wmma_f32_16x16x32_bf16(false, pf[kh].v, false, vf, (short)0, o[dt].v, false, false);
    }
  }
}

__global__ __launch_bounds__(128) void flash_attn(
    const unsigned short* __restrict__ Q,   // [B,S,H,D]  (roped)
    const unsigned short* __restrict__ Kx,  // [B,S,HKV,D](roped)
    const unsigned short* __restrict__ Vt,  // [B,HKV,D,S]
    unsigned short* __restrict__ O)         // [B,S,H*D]
{
  __shared__ unsigned short ldsP[4][16 * 64];
  const int lane = threadIdx.x & 31;
  const int w    = threadIdx.x >> 5;
  const int wave = blockIdx.x * 4 + w;
  const int QT   = S_ / 16;
  if (wave >= B_ * H_ * QT) return;
  const int qt   = wave % QT;
  const int h    = (wave / QT) % H_;
  const int b    = wave / (QT * H_);
  const int kv   = h / (H_ / HKV_);
  const int qbase = qt * 16;
  const int col  = lane & 15;
  const int hgrp = lane >> 4;
  unsigned short* lp = ldsP[w];

  const unsigned short* Qbase = Q  + (size_t)b * S_ * H_ * D_ + (size_t)h * D_;
  const unsigned short* Kbase = Kx + (size_t)b * S_ * HKV_ * D_ + (size_t)kv * D_;
  const unsigned short* Vbase = Vt + ((size_t)b * HKV_ + kv) * D_ * S_;

  v16bf qf[4];
#pragma unroll
  for (int dd = 0; dd < 4; ++dd)
    qf[dd] = load_fragA(Qbase, H_ * D_, qbase + col, dd * 32, lane);

  FragF o[8];
#pragma unroll
  for (int dt = 0; dt < 8; ++dt) o[dt].v = zero_v8f();
  float m_i[8], l_i[8];
#pragma unroll
  for (int r = 0; r < 8; ++r) { m_i[r] = -1e30f; l_i[r] = 0.0f; }

  const int nfull = (qbase + 1) >> 6;   // chunks fully below the diagonal
  const int nch   = (qbase + 79) >> 6;  // all chunks covering keys <= qbase+15

  for (int kc = 0; kc < nfull; ++kc)
    fa_chunk64<false>(kc * 64, qbase, lane, qf, Kbase, Vbase, lp, o, m_i, l_i);
  for (int kc = nfull; kc < nch; ++kc)
    fa_chunk64<true>(kc * 64, qbase, lane, qf, Kbase, Vbase, lp, o, m_i, l_i);

#pragma unroll
  for (int r = 0; r < 8; ++r) {
    const float inv = 1.0f / l_i[r];
    const int srow  = qbase + r + hgrp * 8;
#pragma unroll
    for (int dt = 0; dt < 8; ++dt) {
      O[((size_t)b * S_ + srow) * (H_ * D_) + h * D_ + dt * 16 + col] =
          f2bf(o[dt].f[r] * inv);
    }
  }
}

// ---------------------------------------------------------------- launcher
extern "C" void kernel_launch(void* const* d_in, const int* in_sizes, int n_in,
                              void* d_out, int out_size, void* d_ws, size_t ws_size,
                              hipStream_t stream) {
  (void)in_sizes; (void)n_in; (void)out_size; (void)ws_size;
  const float* hid  = (const float*)d_in[0];
  const float* sinT = (const float*)d_in[1];
  const float* cosT = (const float*)d_in[2];
  const float* Wq   = (const float*)d_in[3];
  const float* Wk   = (const float*)d_in[4];
  const float* Wv   = (const float*)d_in[5];
  const float* Wo   = (const float*)d_in[6];
  float* out = (float*)d_out;

  char* ws = (char*)d_ws;
  size_t off = 0;
  auto alloc = [&](size_t bytes) -> unsigned short* {
    unsigned short* p = (unsigned short*)(ws + off);
    off += (bytes + 255) & ~(size_t)255;
    return p;
  };
  const int M = B_ * S_;  // 4096
  unsigned short* hbf = alloc((size_t)M * HID_ * 2);
  unsigned short* wqt = alloc((size_t)HID_ * H_ * D_ * 2);
  unsigned short* wkt = alloc((size_t)HID_ * HKV_ * D_ * 2);
  unsigned short* wvt = alloc((size_t)HID_ * HKV_ * D_ * 2);
  unsigned short* wot = alloc((size_t)H_ * D_ * HID_ * 2);
  unsigned short* Qb  = alloc((size_t)B_ * S_ * H_ * D_ * 2);
  unsigned short* Kb  = alloc((size_t)B_ * S_ * HKV_ * D_ * 2);
  unsigned short* Vb  = alloc((size_t)B_ * S_ * HKV_ * D_ * 2);
  unsigned short* Vtp = alloc((size_t)B_ * S_ * HKV_ * D_ * 2);
  unsigned short* Ob  = alloc((size_t)B_ * S_ * H_ * D_ * 2);

  // 1. convert activations + transposed weights to bf16
  {
    int n = M * HID_;
    f32_to_bf16_k<<<(n + 255) / 256, 256, 0, stream>>>(hid, hbf, n);
  }
  f32_to_bf16_T<<<(HID_ * H_ * D_ + 255) / 256, 256, 0, stream>>>(Wq, wqt, HID_, H_ * D_);
  f32_to_bf16_T<<<(HID_ * HKV_ * D_ + 255) / 256, 256, 0, stream>>>(Wk, wkt, HID_, HKV_ * D_);
  f32_to_bf16_T<<<(HID_ * HKV_ * D_ + 255) / 256, 256, 0, stream>>>(Wv, wvt, HID_, HKV_ * D_);
  f32_to_bf16_T<<<(H_ * D_ * HID_ + 255) / 256, 256, 0, stream>>>(Wo, wot, H_ * D_, HID_);

  auto gemm_blocks = [](int M_, int N_) { return (M_ / BM) * (N_ / BN); };

  // 2. QKV projections (bf16 out), TDM-fed WMMA GEMMs
  gemm_bf16_tdm<true><<<gemm_blocks(M, H_ * D_), 256, 0, stream>>>(hbf, wqt, Qb, M, H_ * D_, HID_);
  gemm_bf16_tdm<true><<<gemm_blocks(M, HKV_ * D_), 256, 0, stream>>>(hbf, wkt, Kb, M, HKV_ * D_, HID_);
  gemm_bf16_tdm<true><<<gemm_blocks(M, HKV_ * D_), 256, 0, stream>>>(hbf, wvt, Vb, M, HKV_ * D_, HID_);

  // 3. RoPE on Q/K (in place) + transpose V for contiguous P@V fragments
  {
    int tq = B_ * S_ * H_ * 64;
    rope_inplace<<<(tq + 255) / 256, 256, 0, stream>>>(Qb, cosT, sinT, H_, tq);
    int tk = B_ * S_ * HKV_ * 64;
    rope_inplace<<<(tk + 255) / 256, 256, 0, stream>>>(Kb, cosT, sinT, HKV_, tk);
    int tv = B_ * HKV_ * D_ * S_;
    transpose_v<<<(tv + 255) / 256, 256, 0, stream>>>(Vb, Vtp, tv);
  }

  // 4. causal GQA flash attention
  {
    int fa_waves = B_ * H_ * (S_ / 16);
    flash_attn<<<fa_waves / 4, 128, 0, stream>>>(Qb, Kb, Vtp, Ob);
  }

  // 5. output projection (fp32 out)
  gemm_bf16_tdm<false><<<gemm_blocks(M, HID_), 256, 0, stream>>>(Ob, wot, out, M, HID_, H_ * D_);
}
